// MultiheadAttention_40209483825966
// MI455X (gfx1250) — matrix operand order repaired and checked by reading, not compiled, not run on hardware
//
#include <hip/hip_runtime.h>
#include <math.h>

// ---------------------------------------------------------------------------
// MultiheadAttention forward for MI455X (gfx1250), wave32 + bf16 WMMA.
// T=2048, B=1, E=2048, H=32, HD=64.
// GEMM: double-buffered async global->LDS (ASYNCcnt counted waits) feeding
// v_wmma_f32_16x16x32_bf16; all operands pre-converted to bf16 once.
// ---------------------------------------------------------------------------

#define T_DIM    2048
#define E_DIM    2048
#define H_DIM    32
#define HD_DIM   64
#define HALF_DIM 32

#define BM 128
#define BN 128
#define BK 64
#define LDT 72   // padded LDS row stride in ushorts (144B, keeps 16B alignment)

typedef unsigned short ushort_t;
typedef __attribute__((ext_vector_type(16))) __bf16 v16bf;
typedef __attribute__((ext_vector_type(8)))  float  v8f;

// exact operand types for the async-LDS builtin (param0: 'int __vector(4) __device__*')
typedef int v4i_vs __attribute__((vector_size(4 * sizeof(int))));
typedef __attribute__((address_space(1))) v4i_vs* g_v4i_ptr;
typedef __attribute__((address_space(3))) v4i_vs* l_v4i_ptr;

#if defined(__HIP_DEVICE_COMPILE__) && defined(__has_builtin)
#  if __has_builtin(__builtin_amdgcn_global_load_async_to_lds_b128)
#    define HAVE_ASYNC_LDS 1
#  endif
#endif
#ifndef HAVE_ASYNC_LDS
#  define HAVE_ASYNC_LDS 0
#endif

__device__ __forceinline__ unsigned short f2bf(float f) {
    unsigned int u = __float_as_uint(f);
    unsigned int r = (u + 0x7FFFu + ((u >> 16) & 1u)) >> 16;   // round-nearest-even
    return (unsigned short)r;
}

// 16-byte global -> LDS copy; async (ASYNCcnt-tracked) path when available.
__device__ __forceinline__ void copy16_g2l(const ushort_t* __restrict__ g,
                                           ushort_t* __restrict__ l) {
#if HAVE_ASYNC_LDS
    __builtin_amdgcn_global_load_async_to_lds_b128(
        (g_v4i_ptr)(unsigned long long)g,
        (l_v4i_ptr)(unsigned int)(unsigned long long)l,
        0, 0);
#else
    *reinterpret_cast<uint4*>(l) = *reinterpret_cast<const uint4*>(g);
#endif
}

template <int N>
__device__ __forceinline__ void wait_g2l() {
#if HAVE_ASYNC_LDS
#  if __has_builtin(__builtin_amdgcn_s_wait_asynccnt)
    __builtin_amdgcn_s_wait_asynccnt(N);
#  else
    asm volatile("s_wait_asynccnt %0" :: "i"(N) : "memory");
#  endif
#endif
}

// ---------------------------------------------------------------------------
// C[M,N] = A[M,K] * B[N,K]^T (+ bias[N]).  A,B bf16 in memory, fp32 accum.
// 256 threads = 8 waves (2M x 4N); wave strip 64x32 = 4x2 accumulators;
// 16 WMMAs per 64-deep tile; double-buffered async LDS fill: tile k+1
// streams in (8 async issues/wave) while tile k is consumed
// (s_wait_asynccnt 8 exploits in-order async completion).
// causal!=0 skips tiles strictly above the diagonal.
// OOB B rows (N < BN case) are clamped to row N-1: row n only feeds output
// column n, and columns >= N are never stored, so clamping is harmless and
// keeps EXEC full + async issue count uniform.
// ---------------------------------------------------------------------------
__global__ __launch_bounds__(256)
void gemm_abt_wmma(const ushort_t* __restrict__ A, int lda,
                   const ushort_t* __restrict__ B, int ldb,
                   const float* __restrict__ bias,
                   float* __restrict__ C, int ldc,
                   int M, int N, int K, int causal)
{
    __shared__ ushort_t As[2][BM][LDT];
    __shared__ ushort_t Bs[2][BN][LDT];

    const int n0 = blockIdx.x * BN;
    const int m0 = blockIdx.y * BM;
    if (causal && n0 > m0 + BM - 1) return;   // block-uniform: EXEC stays all-1s

    const int tid  = threadIdx.x;
    const int wave = tid >> 5;
    const int lane = tid & 31;
    const int lr   = lane & 15;
    const int hi   = lane >> 4;      // lane-half select
    const int wm   = wave >> 2;      // 0..1  (M strip)
    const int wn   = wave & 3;       // 0..3  (N strip)

    // per-tile fill: 128 rows x 64 bf16 = 8 x 16B chunks per row,
    // 2048 chunks / 256 threads = 4 A + 4 B issues per thread (8 per wave slot)
    auto issue_tile = [&](int k0, int buf) {
#pragma unroll
        for (int it = 0; it < 4; it++) {
            int c   = tid + it * 256;
            int row = c >> 3, c8 = c & 7;
            copy16_g2l(&A[(size_t)(m0 + row) * lda + k0 + c8 * 8],
                       &As[buf][row][c8 * 8]);
        }
#pragma unroll
        for (int it = 0; it < 4; it++) {
            int c   = tid + it * 256;
            int row = c >> 3, c8 = c & 7;
            int rg  = n0 + row;
            rg = rg < N ? rg : N - 1;          // clamp: keeps EXEC full
            copy16_g2l(&B[(size_t)rg * ldb + k0 + c8 * 8],
                       &Bs[buf][row][c8 * 8]);
        }
    };

    v8f acc[4][2];
#pragma unroll
    for (int i = 0; i < 4; i++)
#pragma unroll
        for (int j = 0; j < 2; j++)
            acc[i][j] = (v8f){0.f,0.f,0.f,0.f,0.f,0.f,0.f,0.f};

    const int nk = K / BK;
    issue_tile(0, 0);

    for (int kt = 0; kt < nk; ++kt) {
        const int buf = kt & 1;
        if (kt + 1 < nk) {
            issue_tile((kt + 1) * BK, buf ^ 1);   // prefetch next tile
            wait_g2l<8>();                        // tile kt resident (in-order)
        } else {
            wait_g2l<0>();
        }
        __syncthreads();

        // ---- two 32-deep WMMA sub-steps over the 64-deep tile --------------
#pragma unroll
        for (int ks = 0; ks < BK; ks += 32) {
            // A 16x32 bf16: lane-half hi -> K chunks {hi*8..+7, hi*8+16..+23}
            v16bf afrag[4];
#pragma unroll
            for (int i = 0; i < 4; i++) {
                int row = wm * 64 + i * 16 + lr;
                int kb  = ks + hi * 8;
                union { v16bf v; float4 f[2]; } u;
                u.f[0] = *reinterpret_cast<const float4*>(&As[buf][row][kb]);
                u.f[1] = *reinterpret_cast<const float4*>(&As[buf][row][kb + 16]);
                afrag[i] = u.v;
            }
            // B 32x16 bf16: lane-half hi -> K chunks {hi*8..+7, hi*8+16..+23}
            v16bf bfrag[2];
#pragma unroll
            for (int j = 0; j < 2; j++) {
                int row = wn * 32 + j * 16 + lr;
                int kb  = ks + hi * 8;
                union { v16bf v; float4 f[2]; } u;
                u.f[0] = *reinterpret_cast<const float4*>(&Bs[buf][row][kb]);
                u.f[1] = *reinterpret_cast<const float4*>(&Bs[buf][row][kb + 16]);
                bfrag[j] = u.v;
            }
#pragma unroll
            for (int i = 0; i < 4; i++)
#pragma unroll
                for (int j = 0; j < 2; j++)
                    acc[i][j] = __builtin_amdgcn_wmma_f32_16x16x32_bf16(
                        false, afrag[i], false, bfrag[j],
                        (short)0, acc[i][j], false, false);
        }
        __syncthreads();   // protects buf from next prefetch overwrite
    }

    // ---- store C (VGPR r -> row r (lanes 0-15) / r+8 (lanes 16-31)) --------
#pragma unroll
    for (int i = 0; i < 4; i++) {
        int rbase = m0 + wm * 64 + i * 16 + hi * 8;
#pragma unroll
        for (int j = 0; j < 2; j++) {
            int col = n0 + wn * 32 + j * 16 + lr;
            if (col < N) {
                float bval = bias ? bias[col] : 0.0f;
#pragma unroll
                for (int r = 0; r < 8; r++)
                    C[(size_t)(rbase + r) * ldc + col] = acc[i][j][r] + bval;
            }
        }
    }
}

// ---------------------------------------------------------------------------
// fp32 -> bf16 bulk conversion, 4 elements / thread.
// ---------------------------------------------------------------------------
__global__ void conv_bf16(const float* __restrict__ src,
                          ushort_t* __restrict__ dst, int n4)
{
    int i = blockIdx.x * blockDim.x + threadIdx.x;
    if (i >= n4) return;
    float4 f = reinterpret_cast<const float4*>(src)[i];
    uint2 o;
    o.x = ((unsigned)f2bf(f.y) << 16) | f2bf(f.x);
    o.y = ((unsigned)f2bf(f.w) << 16) | f2bf(f.z);
    reinterpret_cast<uint2*>(dst)[i] = o;
}

// ---------------------------------------------------------------------------
// Rotary embedding: read fp32 (T,E), write bf16 (T,E).
//  k path: scale = 1/xscale ;  q path: scale = xscale then
//  HD^-0.5 * max(1, log(t)/log(1024))   (src_len 2048 > scale_length 1024).
// ---------------------------------------------------------------------------
__global__ void rotary_kernel(const float* __restrict__ x,
                              ushort_t* __restrict__ xb,
                              const float* __restrict__ sinp,
                              const float* __restrict__ cosp,
                              const float* __restrict__ xscale,
                              int is_q)
{
    int p = blockIdx.x * blockDim.x + threadIdx.x;       // pair index
    if (p >= T_DIM * E_DIM / 2) return;
    int t = p / (E_DIM / 2);
    int c = p % (E_DIM / 2);
    int i = c & (HALF_DIM - 1);                          // pair index within head

    float sc = xscale[t * HALF_DIM + i];
    if (!is_q) sc = 1.0f / sc;
    float s  = sinp[t * HALF_DIM + i] * sc;
    float co = cosp[t * HALF_DIM + i] * sc;

    size_t idx = (size_t)t * E_DIM + 2 * c;
    float x0 = x[idx], x1 = x[idx + 1];
    float y0 = x0 * co - x1 * s;
    float y1 = x1 * co + x0 * s;

    if (is_q) {
        // log(t)/log(1024); t==0 -> logf(0)=-inf -> fmaxf picks 1
        float extra = 0.125f * fmaxf(1.0f, logf((float)t) * 0.14426950408889634f);
        y0 *= extra; y1 *= extra;
    }
    reinterpret_cast<unsigned*>(xb)[p] =
        ((unsigned)f2bf(y1) << 16) | f2bf(y0);
}

// v fp32 (T,E) -> vT bf16 (E,T): vT[h*64+d][t] = v[t][h*64+d]
__global__ void transpose_v(const float* __restrict__ v,
                            ushort_t* __restrict__ vTb)
{
    size_t idx = (size_t)blockIdx.x * blockDim.x + threadIdx.x;
    if (idx >= (size_t)T_DIM * E_DIM) return;
    int t = (int)(idx % T_DIM);          // fast dim: coalesced writes
    int e = (int)(idx / T_DIM);
    vTb[(size_t)e * T_DIM + t] = f2bf(v[(size_t)t * E_DIM + e]);
}

// Causal softmax: one 256-thread block per query row t; reads fp32 scores S,
// writes bf16 probs P (zeros for s>t so P is a dense valid GEMM operand).
__global__ __launch_bounds__(256)
void softmax_causal(const float* __restrict__ S, ushort_t* __restrict__ P)
{
    __shared__ float red[256];
    const int t   = blockIdx.x;
    const int tid = threadIdx.x;
    const float* srow = S + (size_t)t * T_DIM;
    ushort_t*    prow = P + (size_t)t * T_DIM;
    const int len = t + 1;

    float m = -3.0e38f;
    for (int s = tid; s < len; s += 256) m = fmaxf(m, srow[s]);
    red[tid] = m; __syncthreads();
    for (int o = 128; o > 0; o >>= 1) {
        if (tid < o) red[tid] = fmaxf(red[tid], red[tid + o]);
        __syncthreads();
    }
    m = red[0]; __syncthreads();

    float sum = 0.0f;
    for (int s = tid; s < len; s += 256) sum += __expf(srow[s] - m);
    red[tid] = sum; __syncthreads();
    for (int o = 128; o > 0; o >>= 1) {
        if (tid < o) red[tid] += red[tid + o];
        __syncthreads();
    }
    const float inv = 1.0f / red[0];

    for (int s = tid; s < len; s += 256) prow[s] = f2bf(__expf(srow[s] - m) * inv);
    for (int s = len + tid; s < T_DIM; s += 256) prow[s] = 0;
}

// ---------------------------------------------------------------------------
extern "C" void kernel_launch(void* const* d_in, const int* in_sizes, int n_in,
                              void* d_out, int out_size, void* d_ws, size_t ws_size,
                              hipStream_t stream)
{
    const float* query  = (const float*)d_in[0];
    const float* key    = (const float*)d_in[1];
    const float* value  = (const float*)d_in[2];
    const float* Wq     = (const float*)d_in[3];
    const float* bq     = (const float*)d_in[4];
    const float* Wk     = (const float*)d_in[5];
    const float* bk     = (const float*)d_in[6];
    const float* Wv     = (const float*)d_in[7];
    const float* bv     = (const float*)d_in[8];
    const float* Wo     = (const float*)d_in[9];
    const float* bo     = (const float*)d_in[10];
    const float* sinp   = (const float*)d_in[11];
    const float* cosp   = (const float*)d_in[12];
    const float* xscale = (const float*)d_in[13];
    // d_in[14] = attn_mask (causal -1e9; applied analytically), d_in[15] = 1024
    float* out = (float*)d_out;

    const size_t TE = (size_t)T_DIM * E_DIM;
    const size_t TT = (size_t)T_DIM * T_DIM;   // == TE here
    float* ws = (float*)d_ws;

    // fp32 region (3 * TE floats), recycled by lifetime:
    float* q    = ws;            // proj q  -> (dead after rotary) -> S (scores)
    float* k    = ws + TE;       // proj k  -> (dead after rotary) -> attn out
    float* v    = ws + 2 * TE;   // proj v  -> (dead after transpose) -> Pb+attnb
    float* S    = q;
    float* attn = k;
    ushort_t* Pb    = (ushort_t*)v;        // TT bf16 probs
    ushort_t* attnb = (ushort_t*)v + TT;   // TE bf16 pre-Wo activations

    // bf16 region (7 * TE ushorts):
    ushort_t* wsb = (ushort_t*)(ws + 3 * TE);
    ushort_t* Wqb = wsb + 0 * TE;
    ushort_t* Wkb = wsb + 1 * TE;
    ushort_t* Wvb = wsb + 2 * TE;
    ushort_t* Wob = wsb + 3 * TE;
    ushort_t* Xqb = wsb + 4 * TE;   // recycled as qb after q-projection
    ushort_t* Xkb = wsb + 5 * TE;   // recycled as kb after k-projection
    ushort_t* Xvb = wsb + 6 * TE;   // recycled as vTb after v-projection
    ushort_t* qb  = Xqb;
    ushort_t* kb  = Xkb;
    ushort_t* vTb = Xvb;

    dim3 blk(256);
    dim3 gproj(E_DIM / BN, T_DIM / BM);
    const int n4 = (int)(TE / 4);
    const int gcv = (n4 + 255) / 256;

    // one-time bf16 conversions of all GEMM operands
    conv_bf16<<<gcv, blk, 0, stream>>>(query, Xqb, n4);
    conv_bf16<<<gcv, blk, 0, stream>>>(key,   Xkb, n4);
    conv_bf16<<<gcv, blk, 0, stream>>>(value, Xvb, n4);
    conv_bf16<<<gcv, blk, 0, stream>>>(Wq, Wqb, n4);
    conv_bf16<<<gcv, blk, 0, stream>>>(Wk, Wkb, n4);
    conv_bf16<<<gcv, blk, 0, stream>>>(Wv, Wvb, n4);
    conv_bf16<<<gcv, blk, 0, stream>>>(Wo, Wob, n4);

    // Q/K/V projections: x @ W^T + b  (fp32 out)
    gemm_abt_wmma<<<gproj, blk, 0, stream>>>(Xqb, E_DIM, Wqb, E_DIM, bq, q, E_DIM,
                                             T_DIM, E_DIM, E_DIM, 0);
    gemm_abt_wmma<<<gproj, blk, 0, stream>>>(Xkb, E_DIM, Wkb, E_DIM, bk, k, E_DIM,
                                             T_DIM, E_DIM, E_DIM, 0);
    gemm_abt_wmma<<<gproj, blk, 0, stream>>>(Xvb, E_DIM, Wvb, E_DIM, bv, v, E_DIM,
                                             T_DIM, E_DIM, E_DIM, 0);

    const int nrot = T_DIM * E_DIM / 2;
    rotary_kernel<<<(nrot + 255) / 256, blk, 0, stream>>>(q, qb, sinp, cosp, xscale, 1);
    rotary_kernel<<<(nrot + 255) / 256, blk, 0, stream>>>(k, kb, sinp, cosp, xscale, 0);

    transpose_v<<<(unsigned)((TE + 255) / 256), blk, 0, stream>>>(v, vTb);

    for (int h = 0; h < H_DIM; h++) {
        // S = q_h @ k_h^T  (M=N=T, K=64), upper-triangle tiles skipped
        gemm_abt_wmma<<<dim3(T_DIM / BN, T_DIM / BM), blk, 0, stream>>>(
            qb + h * HD_DIM, E_DIM, kb + h * HD_DIM, E_DIM, nullptr, S, T_DIM,
            T_DIM, T_DIM, HD_DIM, 1);
        softmax_causal<<<T_DIM, blk, 0, stream>>>(S, Pb);
        // attn[:, h*64 : h*64+64] = P @ v_h
        gemm_abt_wmma<<<dim3(1, T_DIM / BM), blk, 0, stream>>>(
            Pb, T_DIM, vTb + (size_t)h * HD_DIM * T_DIM, T_DIM, nullptr,
            attn + h * HD_DIM, E_DIM,
            T_DIM, HD_DIM, T_DIM, 0);
    }

    // Output projection: out = attn @ Wo^T + bo
    conv_bf16<<<gcv, blk, 0, stream>>>(attn, attnb, n4);
    gemm_abt_wmma<<<gproj, blk, 0, stream>>>(attnb, E_DIM, Wob, E_DIM, bo, out, E_DIM,
                                             T_DIM, E_DIM, E_DIM, 0);
}
